// AttentiveFPModule_9345848836583
// MI455X (gfx1250) — matrix operand order repaired and checked by reading, not compiled, not run on hardware
//
#include <hip/hip_runtime.h>
#include <stdint.h>

// ---------------------------------------------------------------------------
// Problem constants (from the reference)
// ---------------------------------------------------------------------------
#define NN 100000
#define NE 400000
#define NB 4096
#define FN 78
#define FE 11
#define GG 256
#define G3 768

typedef __attribute__((ext_vector_type(16))) __bf16 v16bf;
typedef __attribute__((ext_vector_type(8)))  float  v8f;

// A-loader modes
#define AM_DIR 0
#define AM_ELU 1
#define AM_CAT 2
// Epilogue modes
#define EP_LRELU 0
#define EP_PLAIN 1
#define EP_SCAT  2

__device__ __forceinline__ unsigned pk2bf(float a, float b) {
    union { __bf16 h[2]; unsigned u; } x;
    x.h[0] = (__bf16)a;
    x.h[1] = (__bf16)b;
    return x.u;
}
__device__ __forceinline__ unsigned short s2bf(float a) {
    union { __bf16 h; unsigned short u; } x;
    x.h = (__bf16)a;
    return x.u;
}
__device__ __forceinline__ float eluf(float x) {
    return (x > 0.0f) ? x : (__expf(x) - 1.0f);
}

// order-preserving float <-> uint key (for atomicMax-based segment max)
__device__ __forceinline__ unsigned fkey(float f) {
    union { float f; unsigned u; } x; x.f = f;
    unsigned u = x.u;
    return (u & 0x80000000u) ? ~u : (u | 0x80000000u);
}
__device__ __forceinline__ float finv(unsigned k) {
    unsigned u = (k & 0x80000000u) ? (k ^ 0x80000000u) : ~k;
    union { unsigned u; float f; } x; x.u = u;
    return x.f;
}

// ---------------------------------------------------------------------------
// bf16 WMMA GEMM:  out[M,Nout] = epi( A'[M,K] @ W[Nout,K]^T + bias )
//   A' per AMODE: direct fp32 rows, ELU(fp32 rows), or gathered concat
//   (node_feats[gidx[row]] ++ edge_feats[row]).
//   EPI: leaky-relu store, plain store, or per-row-scaled atomic scatter:
//        atomicAdd(out[sidx[row]*Nout + col], sscale[row] * (acc+bias)).
//   VEC (compile-time): K % 32 == 0 -> each thread stages one 8-element
//   chunk with two b128 global loads + one b128 LDS store of packed bf16
//   (no in-range guards needed). !VEC -> scalar staging with zero padding.
// Tile 64x64x32, 256 threads = 8 waves, each wave owns a 16x32 strip
// (two v_wmma_f32_16x16x32_bf16 per K-step).
// ---------------------------------------------------------------------------
template <int AMODE, int EPI, bool VEC>
__global__ __launch_bounds__(256)
void gemm_wmma_kernel(const float* __restrict__ A, const float* __restrict__ A2,
                      const int* __restrict__ gidx,
                      const float* __restrict__ W, const float* __restrict__ bias,
                      float* __restrict__ out,
                      const int* __restrict__ sidx, const float* __restrict__ sscale,
                      int M, int K, int Nout, int F1, int F2)
{
    static_assert(!(VEC && AMODE == AM_CAT), "gather-concat uses scalar staging");

    __shared__ alignas(16) unsigned short As[64 * 40];  // [m][k], stride 40 (pad)
    __shared__ alignas(16) unsigned short Bs[64 * 40];  // [n][k], stride 40 (pad)

    const int tid  = threadIdx.x;
    const int n0   = blockIdx.x * 64;
    const int m0   = blockIdx.y * 64;
    const int w    = tid >> 5;
    const int lane = tid & 31;
    const int wr   = w & 3;        // 16-row block within tile
    const int wc   = w >> 2;       // 32-col block within tile
    const int ml   = lane & 15;
    const int hi   = lane >> 4;

    // chunk assignment for vectorized staging: 64 rows x 4 chunks of 8
    const int cm  = tid >> 2;           // row within tile
    const int ckb = (tid & 3) * 8;      // k-offset within tile (0,8,16,24)

    v8f acc0 = {};
    v8f acc1 = {};

    for (int k0 = 0; k0 < K; k0 += 32) {
        if constexpr (VEC) {
            // ---- vector stage A: two float4 loads -> 8 packed bf16 (16B) ----
            {
                int row = m0 + cm, gk = k0 + ckb;
                float4 x0 = {0, 0, 0, 0}, x1 = {0, 0, 0, 0};
                if (row < M) {
                    const float* p = A + (size_t)row * K + gk;
                    x0 = *(const float4*)p;
                    x1 = *(const float4*)(p + 4);
                    if (gk + 32 < K) __builtin_prefetch(p + 32, 0, 1);
                }
                if constexpr (AMODE == AM_ELU) {
                    x0.x = eluf(x0.x); x0.y = eluf(x0.y); x0.z = eluf(x0.z); x0.w = eluf(x0.w);
                    x1.x = eluf(x1.x); x1.y = eluf(x1.y); x1.z = eluf(x1.z); x1.w = eluf(x1.w);
                }
                uint4 u;
                u.x = pk2bf(x0.x, x0.y); u.y = pk2bf(x0.z, x0.w);
                u.z = pk2bf(x1.x, x1.y); u.w = pk2bf(x1.z, x1.w);
                *(uint4*)&As[cm * 40 + ckb] = u;   // byte addr = cm*80 + ckb*2 : 16B aligned
            }
            // ---- vector stage B: Bs[n][k] = W[n0+n][k0+k] ----
            {
                int col = n0 + cm, gk = k0 + ckb;
                float4 x0 = {0, 0, 0, 0}, x1 = {0, 0, 0, 0};
                if (col < Nout) {
                    const float* p = W + (size_t)col * K + gk;
                    x0 = *(const float4*)p;
                    x1 = *(const float4*)(p + 4);
                }
                uint4 u;
                u.x = pk2bf(x0.x, x0.y); u.y = pk2bf(x0.z, x0.w);
                u.z = pk2bf(x1.x, x1.y); u.w = pk2bf(x1.z, x1.w);
                *(uint4*)&Bs[cm * 40 + ckb] = u;
            }
        } else {
            // ---- scalar stage A (fused transform / gather-concat, zero pad) ----
            for (int i = tid; i < 64 * 32; i += 256) {
                int m = i >> 5, k = i & 31;
                int row = m0 + m, gk = k0 + k;
                float v = 0.0f;
                if (row < M && gk < K) {
                    if constexpr (AMODE == AM_DIR) {
                        v = A[(size_t)row * K + gk];
                    } else if constexpr (AMODE == AM_ELU) {
                        v = eluf(A[(size_t)row * K + gk]);
                    } else {  // AM_CAT
                        if (gk < F1) v = A[(size_t)gidx[row] * F1 + gk];
                        else         v = A2[(size_t)row * F2 + (gk - F1)];
                    }
                }
                As[m * 40 + k] = s2bf(v);
            }
            // ---- scalar stage B ----
            for (int i = tid; i < 64 * 32; i += 256) {
                int n = i >> 5, k = i & 31;
                int col = n0 + n, gk = k0 + k;
                float v = (col < Nout && gk < K) ? W[(size_t)col * K + gk] : 0.0f;
                Bs[n * 40 + k] = s2bf(v);
            }
        }
        __syncthreads();

        // ---- build fragments per ISA 16-bit A 16x32 / B 32x16 layouts ----
        union { v16bf v; unsigned u[8]; } af, bf0, bf1;
#pragma unroll
        for (int v = 0; v < 8; ++v) {
            int k2 = ((v < 4) ? (2 * v) : (16 + 2 * (v - 4))) + hi * 8;
            af.u[v]  = *(const unsigned*)&As[(wr * 16 + ml) * 40 + k2];
            int kb = k2;
            bf0.u[v] = *(const unsigned*)&Bs[(wc * 32 + ml) * 40 + kb];
            bf1.u[v] = *(const unsigned*)&Bs[(wc * 32 + 16 + ml) * 40 + kb];
        }
        acc0 = __builtin_amdgcn_wmma_f32_16x16x32_bf16(false, af.v, false, bf0.v,
                                                       (short)0, acc0, false, false);
        acc1 = __builtin_amdgcn_wmma_f32_16x16x32_bf16(false, af.v, false, bf1.v,
                                                       (short)0, acc1, false, false);
        __syncthreads();
    }

    // ---- epilogue (C layout: lane v -> M = v + 8*hi, N = lane&15) ----
#pragma unroll
    for (int v = 0; v < 8; ++v) {
        int row = m0 + wr * 16 + v + hi * 8;
        if (row >= M) continue;
#pragma unroll
        for (int s = 0; s < 2; ++s) {
            int col = n0 + wc * 32 + s * 16 + ml;
            if (col >= Nout) continue;
            float acc = (s == 0) ? acc0[v] : acc1[v];
            float val = acc + bias[col];
            if constexpr (EPI == EP_LRELU) {
                out[(size_t)row * Nout + col] = (val > 0.0f) ? val : 0.01f * val;
            } else if constexpr (EPI == EP_PLAIN) {
                out[(size_t)row * Nout + col] = val;
            } else {
                atomicAdd(&out[(size_t)sidx[row] * Nout + col], val * sscale[row]);
            }
        }
    }
}

// ---------------------------------------------------------------------------
// Attention-logit dot products: out[e] = lrelu(w[0:256].X[ia[e]] +
//                                              w[256:512].Y[ib[e]] + b)
// One wave32 per item, shuffle reduction. ia/ib == nullptr -> identity.
// ---------------------------------------------------------------------------
__global__ __launch_bounds__(256)
void pair_dot_kernel(const float* __restrict__ X, const float* __restrict__ Y,
                     const int* __restrict__ ia, const int* __restrict__ ib,
                     const float* __restrict__ w, const float* __restrict__ bptr,
                     float* __restrict__ out, int E, int relu_x)
{
    int e = blockIdx.x * 8 + (threadIdx.x >> 5);
    if (e >= E) return;
    int lane = threadIdx.x & 31;
    int ra = ia ? ia[e] : e;
    int rb = ib ? ib[e] : e;
    const float* xr = X + (size_t)ra * GG;
    const float* yr = Y + (size_t)rb * GG;
    float s = 0.0f;
#pragma unroll
    for (int k = lane; k < GG; k += 32) {
        float xv = xr[k];
        if (relu_x) xv = (xv > 0.0f) ? xv : 0.0f;
        s += w[k] * xv + w[GG + k] * yr[k];
    }
#pragma unroll
    for (int o = 16; o; o >>= 1) s += __shfl_xor(s, o, 32);
    if (lane == 0) {
        float v = s + bptr[0];
        out[e] = (v > 0.0f) ? v : 0.01f * v;
    }
}

// ---------------------------------------------------------------------------
// Segment softmax pieces
// ---------------------------------------------------------------------------
__global__ void fill_u32_kernel(unsigned* p, unsigned v, int n) {
    int i = blockIdx.x * 256 + threadIdx.x;
    if (i < n) p[i] = v;
}
__global__ void seg_max_kernel(const float* __restrict__ l, const int* __restrict__ seg,
                               unsigned* __restrict__ mk, int n) {
    int i = blockIdx.x * 256 + threadIdx.x;
    if (i < n) atomicMax(&mk[seg[i]], fkey(l[i]));
}
__global__ void seg_exp_kernel(float* __restrict__ l, const int* __restrict__ seg,
                               const unsigned* __restrict__ mk, float* __restrict__ sum, int n) {
    int i = blockIdx.x * 256 + threadIdx.x;
    if (i >= n) return;
    float m = finv(mk[seg[i]]);
    float e = __expf(l[i] - m);
    l[i] = e;
    atomicAdd(&sum[seg[i]], e);
}
__global__ void seg_div_kernel(float* __restrict__ l, const int* __restrict__ seg,
                               const float* __restrict__ sum, int n) {
    int i = blockIdx.x * 256 + threadIdx.x;
    if (i < n) l[i] /= sum[seg[i]];
}

// ---------------------------------------------------------------------------
// c[dst[e]] += a[e] * hv[src[e]]   (one block per edge, 256 feature lanes)
// ---------------------------------------------------------------------------
__global__ __launch_bounds__(256)
void edge_scatter_kernel(const float* __restrict__ hv, const float* __restrict__ a,
                         const int* __restrict__ src, const int* __restrict__ dst,
                         float* __restrict__ c, int E)
{
    int e = blockIdx.x;
    if (e >= E) return;
    int g = threadIdx.x;
    float v = a[e] * hv[(size_t)src[e] * GG + g];
    atomicAdd(&c[(size_t)dst[e] * GG + g], v);
}

// g_feats[ng[n]] += h[n]
__global__ void node_segsum_kernel(const float* __restrict__ h, const int* __restrict__ ng,
                                   float* __restrict__ gbuf, int N)
{
    int i = blockIdx.x * 256 + threadIdx.x;
    if (i >= N * GG) return;
    int r = i >> 8, g = i & 255;
    atomicAdd(&gbuf[(size_t)ng[r] * GG + g], h[i]);
}

// ---------------------------------------------------------------------------
// GRU gate combine, in place on h:  h = relu(gru(gi, gh, h))
// ---------------------------------------------------------------------------
__global__ void gru_combine_kernel(const float* __restrict__ gi, const float* __restrict__ gh,
                                   float* __restrict__ h, int M)
{
    int i = blockIdx.x * 256 + threadIdx.x;
    if (i >= M * GG) return;
    int r = i >> 8, g = i & 255;
    const float* gir = gi + (size_t)r * G3;
    const float* ghr = gh + (size_t)r * G3;
    float ir = gir[g],          hr = ghr[g];
    float iz = gir[GG + g],     hz = ghr[GG + g];
    float in = gir[2 * GG + g], hn = ghr[2 * GG + g];
    float rr = 1.0f / (1.0f + __expf(-(ir + hr)));
    float zz = 1.0f / (1.0f + __expf(-(iz + hz)));
    float nn = tanhf(in + rr * hn);
    float o  = (1.0f - zz) * nn + zz * h[i];
    h[i] = (o > 0.0f) ? o : 0.0f;
}

// ---------------------------------------------------------------------------
// LayerNorm over last dim (256), one block per graph, writes d_out
// ---------------------------------------------------------------------------
__global__ __launch_bounds__(256)
void layernorm_kernel(const float* __restrict__ g, const float* __restrict__ gamma,
                      const float* __restrict__ beta, float* __restrict__ out)
{
    __shared__ float red[256];
    int r = blockIdx.x, t = threadIdx.x;
    float v = g[(size_t)r * GG + t];
    red[t] = v; __syncthreads();
    for (int o = 128; o; o >>= 1) { if (t < o) red[t] += red[t + o]; __syncthreads(); }
    float mu = red[0] * (1.0f / GG);
    __syncthreads();
    float d = v - mu;
    red[t] = d * d; __syncthreads();
    for (int o = 128; o; o >>= 1) { if (t < o) red[t] += red[t + o]; __syncthreads(); }
    float var = red[0] * (1.0f / GG);
    out[(size_t)r * GG + t] = d * rsqrtf(var + 1e-5f) * gamma[t] + beta[t];
}

// ---------------------------------------------------------------------------
// Launch orchestration
// ---------------------------------------------------------------------------
extern "C" void kernel_launch(void* const* d_in, const int* in_sizes, int n_in,
                              void* d_out, int out_size, void* d_ws, size_t ws_size,
                              hipStream_t stream)
{
    (void)in_sizes; (void)n_in; (void)out_size;

    const float* node_feats = (const float*)d_in[0];
    const float* edge_feats = (const float*)d_in[1];
    const int*   src        = (const int*)d_in[2];
    const int*   dst        = (const int*)d_in[3];
    const int*   node_graph = (const int*)d_in[4];
    const float* W_pn   = (const float*)d_in[5];  const float* b_pn   = (const float*)d_in[6];
    const float* W_pe1  = (const float*)d_in[7];  const float* b_pe1  = (const float*)d_in[8];
    const float* W_pe2  = (const float*)d_in[9];  const float* b_pe2  = (const float*)d_in[10];
    const float* W_et   = (const float*)d_in[11]; const float* b_et   = (const float*)d_in[12];
    const float* g0_Wih = (const float*)d_in[13]; const float* g0_Whh = (const float*)d_in[14];
    const float* g0_bih = (const float*)d_in[15]; const float* g0_bhh = (const float*)d_in[16];
    const float* gnn_W_pe = (const float*)d_in[17]; const float* gnn_b_pe = (const float*)d_in[18];
    const float* gnn_W_pn = (const float*)d_in[19]; const float* gnn_b_pn = (const float*)d_in[20];
    const float* gnn_Wih  = (const float*)d_in[21]; const float* gnn_Whh  = (const float*)d_in[22];
    const float* gnn_bih  = (const float*)d_in[23]; const float* gnn_bhh  = (const float*)d_in[24];
    const float* ro_W_cl  = (const float*)d_in[25]; const float* ro_b_cl  = (const float*)d_in[26];
    const float* ro_W_pn  = (const float*)d_in[27]; const float* ro_b_pn  = (const float*)d_in[28];
    const float* ro_Wih   = (const float*)d_in[29]; const float* ro_Whh   = (const float*)d_in[30];
    const float* ro_bih   = (const float*)d_in[31]; const float* ro_bhh   = (const float*)d_in[32];
    const float* ln_gamma = (const float*)d_in[33]; const float* ln_beta  = (const float*)d_in[34];

    // ---- workspace layout (floats) ----
    size_t off_h   = 0;                                   // [N,G]  hv_new / h (in-place GRU)
    size_t off_c   = off_h   + (size_t)NN * GG;           // [N,G]  context accumulator
    size_t off_he1 = off_c   + (size_t)NN * GG;           // [E,G]  he1, reused as hv (N<=E)
    size_t off_gi  = off_he1 + (size_t)NE * GG;           // [N,768]
    size_t off_gh  = off_gi  + (size_t)NN * G3;           // [N,768]
    size_t off_a   = off_gh  + (size_t)NN * G3;           // [E] logits -> e -> a (in place)
    size_t off_an  = off_a   + (size_t)NE;                // [N] readout attention
    size_t off_m   = off_an  + (size_t)NN;                // [N] seg-max keys (uint)
    size_t off_s   = off_m   + (size_t)NN;                // [N] seg sums
    size_t off_g   = off_s   + (size_t)NN;                // [B,G] graph feats
    size_t off_gr  = off_g   + (size_t)NB * GG;           // [B,G] readout context
    size_t total   = off_gr  + (size_t)NB * GG;
    if (ws_size < total * sizeof(float)) return;          // insufficient scratch: no-op

    float*    ws    = (float*)d_ws;
    float*    h     = ws + off_h;
    float*    c     = ws + off_c;
    float*    he1   = ws + off_he1;   // also hv
    float*    gi    = ws + off_gi;
    float*    gh    = ws + off_gh;
    float*    abuf  = ws + off_a;
    float*    anode = ws + off_an;
    unsigned* mk    = (unsigned*)(ws + off_m);
    float*    sbuf  = ws + off_s;
    float*    gbuf  = ws + off_g;
    float*    grbuf = ws + off_gr;

    const unsigned KEY_NEG_INF = 0x007FFFFFu;  // fkey(-inf)
    dim3 blk(256);
    auto gemm_grid = [](int M, int N) { return dim3((unsigned)((N + 63) / 64), (unsigned)((M + 63) / 64)); };
    dim3 gE((NE + 255) / 256), gEw((NE + 7) / 8), gNw((NN + 7) / 8);
    dim3 gNG((NN * GG + 255) / 256), gBG((NB * GG + 255) / 256);

    // ================= GetContext =================
    // hv_new = lrelu(node_feats @ W_pn^T + b_pn) -> h   (K=78: scalar staging)
    gemm_wmma_kernel<AM_DIR, EP_LRELU, false><<<gemm_grid(NN, GG), blk, 0, stream>>>(
        node_feats, nullptr, nullptr, W_pn, b_pn, h, nullptr, nullptr, NN, FN, GG, 0, 0);
    // he1 = lrelu(concat(node_feats[src], edge_feats) @ W_pe1^T + b_pe1)  (K=89)
    gemm_wmma_kernel<AM_CAT, EP_LRELU, false><<<gemm_grid(NE, GG), blk, 0, stream>>>(
        node_feats, edge_feats, src, W_pe1, b_pe1, he1, nullptr, nullptr, NE, FN + FE, GG, FN, FE);
    // logits = lrelu(concat(hv_new[dst], he1) @ W_pe2^T + b)
    pair_dot_kernel<<<gEw, blk, 0, stream>>>(h, he1, dst, nullptr, W_pe2, b_pe2, abuf, NE, 0);
    // edge softmax over dst
    fill_u32_kernel<<<(NN + 255) / 256, blk, 0, stream>>>(mk, KEY_NEG_INF, NN);
    hipMemsetAsync(sbuf, 0, (size_t)NN * sizeof(float), stream);
    seg_max_kernel<<<gE, blk, 0, stream>>>(abuf, dst, mk, NE);
    seg_exp_kernel<<<gE, blk, 0, stream>>>(abuf, dst, mk, sbuf, NE);
    seg_div_kernel<<<gE, blk, 0, stream>>>(abuf, dst, sbuf, NE);
    // c = segsum_dst( a * (he1 @ W_et^T + b_et) )  -- fused scatter epilogue
    hipMemsetAsync(c, 0, (size_t)NN * GG * sizeof(float), stream);
    gemm_wmma_kernel<AM_DIR, EP_SCAT, true><<<gemm_grid(NE, GG), blk, 0, stream>>>(
        he1, nullptr, nullptr, W_et, b_et, c, dst, abuf, NE, GG, GG, 0, 0);
    // h = relu(gru(elu(c), hv_new))
    gemm_wmma_kernel<AM_ELU, EP_PLAIN, true><<<gemm_grid(NN, G3), blk, 0, stream>>>(
        c, nullptr, nullptr, g0_Wih, g0_bih, gi, nullptr, nullptr, NN, GG, G3, 0, 0);
    gemm_wmma_kernel<AM_DIR, EP_PLAIN, true><<<gemm_grid(NN, G3), blk, 0, stream>>>(
        h, nullptr, nullptr, g0_Whh, g0_bhh, gh, nullptr, nullptr, NN, GG, G3, 0, 0);
    gru_combine_kernel<<<gNG, blk, 0, stream>>>(gi, gh, h, NN);

    // ================= GNN layers (AttentiveGRU2) =================
    for (int l = 0; l < 2; ++l) {
        pair_dot_kernel<<<gEw, blk, 0, stream>>>(
            h, h, dst, src, gnn_W_pe + (size_t)l * 2 * GG, gnn_b_pe + l, abuf, NE, 0);
        fill_u32_kernel<<<(NN + 255) / 256, blk, 0, stream>>>(mk, KEY_NEG_INF, NN);
        hipMemsetAsync(sbuf, 0, (size_t)NN * sizeof(float), stream);
        seg_max_kernel<<<gE, blk, 0, stream>>>(abuf, dst, mk, NE);
        seg_exp_kernel<<<gE, blk, 0, stream>>>(abuf, dst, mk, sbuf, NE);
        seg_div_kernel<<<gE, blk, 0, stream>>>(abuf, dst, sbuf, NE);
        // hv = h @ W_pn^T + b  (into he1 buffer)
        gemm_wmma_kernel<AM_DIR, EP_PLAIN, true><<<gemm_grid(NN, GG), blk, 0, stream>>>(
            h, nullptr, nullptr, gnn_W_pn + (size_t)l * GG * GG, gnn_b_pn + (size_t)l * GG,
            he1, nullptr, nullptr, NN, GG, GG, 0, 0);
        // c = segsum_dst(a * hv[src])
        hipMemsetAsync(c, 0, (size_t)NN * GG * sizeof(float), stream);
        edge_scatter_kernel<<<dim3(NE), blk, 0, stream>>>(he1, abuf, src, dst, c, NE);
        // h = relu(gru(elu(c), h))
        gemm_wmma_kernel<AM_ELU, EP_PLAIN, true><<<gemm_grid(NN, G3), blk, 0, stream>>>(
            c, nullptr, nullptr, gnn_Wih + (size_t)l * G3 * GG, gnn_bih + (size_t)l * G3,
            gi, nullptr, nullptr, NN, GG, G3, 0, 0);
        gemm_wmma_kernel<AM_DIR, EP_PLAIN, true><<<gemm_grid(NN, G3), blk, 0, stream>>>(
            h, nullptr, nullptr, gnn_Whh + (size_t)l * G3 * GG, gnn_bhh + (size_t)l * G3,
            gh, nullptr, nullptr, NN, GG, G3, 0, 0);
        gru_combine_kernel<<<gNG, blk, 0, stream>>>(gi, gh, h, NN);
    }

    // ================= AttentiveFPReadout =================
    hipMemsetAsync(gbuf, 0, (size_t)NB * GG * sizeof(float), stream);
    node_segsum_kernel<<<gNG, blk, 0, stream>>>(h, node_graph, gbuf, NN);
    for (int t = 0; t < 2; ++t) {
        // z = lrelu(concat(relu(g)[node_graph], h) @ W_cl^T + b)
        pair_dot_kernel<<<gNw, blk, 0, stream>>>(
            gbuf, h, node_graph, nullptr, ro_W_cl + (size_t)t * 2 * GG, ro_b_cl + t, anode, NN, 1);
        // softmax over graphs
        fill_u32_kernel<<<(NB + 255) / 256, blk, 0, stream>>>(mk, KEY_NEG_INF, NB);
        hipMemsetAsync(sbuf, 0, (size_t)NB * sizeof(float), stream);
        seg_max_kernel<<<dim3((NN + 255) / 256), blk, 0, stream>>>(anode, node_graph, mk, NN);
        seg_exp_kernel<<<dim3((NN + 255) / 256), blk, 0, stream>>>(anode, node_graph, mk, sbuf, NN);
        seg_div_kernel<<<dim3((NN + 255) / 256), blk, 0, stream>>>(anode, node_graph, sbuf, NN);
        // gr = segsum_graph( a * (h @ W_pn^T + b) ) -- fused scatter epilogue
        hipMemsetAsync(grbuf, 0, (size_t)NB * GG * sizeof(float), stream);
        gemm_wmma_kernel<AM_DIR, EP_SCAT, true><<<gemm_grid(NN, GG), blk, 0, stream>>>(
            h, nullptr, nullptr, ro_W_pn + (size_t)t * GG * GG, ro_b_pn + (size_t)t * GG,
            grbuf, node_graph, anode, NN, GG, GG, 0, 0);
        // g = relu(gru(elu(gr), g))
        gemm_wmma_kernel<AM_ELU, EP_PLAIN, true><<<gemm_grid(NB, G3), blk, 0, stream>>>(
            grbuf, nullptr, nullptr, ro_Wih + (size_t)t * G3 * GG, ro_bih + (size_t)t * G3,
            gi, nullptr, nullptr, NB, GG, G3, 0, 0);
        gemm_wmma_kernel<AM_DIR, EP_PLAIN, true><<<gemm_grid(NB, G3), blk, 0, stream>>>(
            gbuf, nullptr, nullptr, ro_Whh + (size_t)t * G3 * GG, ro_bhh + (size_t)t * G3,
            gh, nullptr, nullptr, NB, GG, G3, 0, 0);
        gru_combine_kernel<<<gBG, blk, 0, stream>>>(gi, gh, gbuf, NB);
    }

    // ================= LayerNorm -> d_out =================
    layernorm_kernel<<<dim3(NB), blk, 0, stream>>>(gbuf, ln_gamma, ln_beta, (float*)d_out);
}
